// ServerModelLSTM_30081950941645
// MI455X (gfx1250) — matrix operand order repaired and checked by reading, not compiled, not run on hardware
//
#include <hip/hip_runtime.h>
#include <hip/hip_bf16.h>

#define NL       20
#define T_STEPS  1024
#define BATCH    512
#define IN0      64
#define HH       32
#define G4       128                 // 4*H
#define BTILE    128                 // batch rows per workgroup
#define NSLICE   (BATCH / BTILE)     // 4
#define CT       8                   // timesteps per sync chunk
#define RING_W   32                  // ring window in timesteps (power of 2)
#define RING_CHUNKS (RING_W / CT)    // 4

typedef _Float16 f16;
typedef __attribute__((ext_vector_type(16))) _Float16 v16h;
typedef __attribute__((ext_vector_type(8)))  _Float16 v8h;
typedef __attribute__((ext_vector_type(8)))  float    v8f;

union V16H { v16h v; v8h h[2]; };

// ---- transcendentals: prefer gfx1250 hardware V_TANH_F32 ----
#if __has_builtin(__builtin_amdgcn_tanhf)
#define TANHF(x) __builtin_amdgcn_tanhf(x)
#elif __has_builtin(__builtin_amdgcn_tanh_f32)
#define TANHF(x) __builtin_amdgcn_tanh_f32(x)
#else
__device__ __forceinline__ float __sw_tanh(float x) {
    return 2.0f * __builtin_amdgcn_rcpf(1.0f + __builtin_amdgcn_exp2f(-2.88539008f * x)) - 1.0f;
}
#define TANHF(x) __sw_tanh(x)
#endif
__device__ __forceinline__ float fast_sig(float x) {
    // sigmoid(x) = 0.5*tanh(x/2) + 0.5  (1 trans + fma)
    return __builtin_fmaf(0.5f, TANHF(0.5f * x), 0.5f);
}

// ---- CDNA5 async global->LDS (per-lane 16B), tracked by ASYNCcnt ----
__device__ __forceinline__ void async_ld_b128(unsigned lds_byte_off, const f16* g) {
    asm volatile("global_load_async_to_lds_b128 %0, %1, off"
                 :: "v"(lds_byte_off), "v"(g) : "memory");
}
#define WAIT_ASYNC(n) asm volatile("s_wait_asynccnt %0" :: "i"(n) : "memory")

__device__ __forceinline__ unsigned lds_off_of(const void* p) {
    // generic "shared" address: low 32 bits are the LDS byte offset
    return (unsigned)(unsigned long long)p;
}

// ---------------------------------------------------------------------------
// Pipelined LSTM layer: one WG = (layer, batch-slice), persistent over all T.
// KXC = number of 32-wide K chunks from the input x (2 for layer 0, 1 else);
// the h-recurrence adds one more. XRING: input comes from an L2-resident ring
// written by the previous layer (async double-buffered through LDS).
// ---------------------------------------------------------------------------
template<int KXC, bool XRING>
__device__ void lstm_layer_pipeline(
    const f16* __restrict__ xin,   // layer0: x16 (T,B,IN0); else ring[l-1]
    f16* __restrict__ ring_out,    // (RING_W, BATCH, HH) f16
    f16* __restrict__ hT_out,      // (BATCH, HH) f16 or nullptr
    int layer, int slice,
    int* wprog, int* rprog,
    const float* __restrict__ WihG, const float* __restrict__ WhhG,
    const float* __restrict__ biG,  const float* __restrict__ bhG,
    f16* sW, float* sBias, f16* sH, f16* sX)
{
    const int tid  = threadIdx.x;
    const int wave = tid >> 5;
    const int lane = tid & 31;
    const int l16  = lane & 15;
    const int hi   = lane >> 4;          // K-half / M-half selector
    constexpr int KC  = KXC + 1;         // total 32-wide K chunks
    constexpr int KIN = KXC * 32;        // x K extent

    // ---- stage combined weights to LDS as Wcomb[k][n] (f16) ----
    for (int idx = tid; idx < (KIN + HH) * G4; idx += blockDim.x) {
        int k = idx / G4, n = idx % G4;
        float w = (k < KIN) ? WihG[n * KIN + k] : WhhG[n * HH + (k - KIN)];
        sW[idx] = (f16)w;
    }
    for (int n = tid; n < G4; n += blockDim.x) sBias[n] = biG[n] + bhG[n];
    for (int i = tid; i < BTILE * HH; i += blockDim.x) sH[i] = (f16)0.0f;
    __syncthreads();

    // ---- stationary B-matrix registers: breg[kc][n] covers K=32kc..+31, N=16n..+15
    v16h breg[KC][8];
#pragma unroll
    for (int kc = 0; kc < KC; ++kc) {
#pragma unroll
        for (int n = 0; n < 8; ++n) {
            V16H tmp;
#pragma unroll
            for (int j = 0; j < 16; ++j)
                tmp.v[j] = sW[(kc * 32 + hi * 8 + (j & 7) + (j >> 3) * 16) * G4 + (n * 16 + l16)];
            breg[kc][n] = tmp.v;
        }
    }

    float biasn[8];
#pragma unroll
    for (int n = 0; n < 8; ++n) biasn[n] = sBias[n * 16 + l16];

    // ---- persistent cell state in D-layout registers ----
    v8f cst[2];
#pragma unroll
    for (int j = 0; j < 2; ++j)
#pragma unroll
        for (int r = 0; r < 8; ++r) cst[j][r] = 0.0f;

    const int mrowA    = wave * 16 + l16;        // A-matrix row within slice
    const int growA    = slice * BTILE + mrowA;  // global batch row
    const int rowbaseD = wave * 16 + hi * 8;     // D-matrix row base within slice

    const int myW = layer * NSLICE + slice;
    const int upW = (layer - 1) * NSLICE + slice;

    for (int chunk = 0; chunk < T_STEPS / CT; ++chunk) {
        // ---- acquire: upstream data ready, downstream ring space free ----
        if (tid == 0) {
            if (layer > 0) {
                while (__hip_atomic_load(&wprog[upW], __ATOMIC_ACQUIRE,
                                         __HIP_MEMORY_SCOPE_AGENT) < chunk + 1)
                    __builtin_amdgcn_s_sleep(2);
            }
            int need = chunk - RING_CHUNKS + 1;
            if (need > 0) {
                while (__hip_atomic_load(&rprog[myW], __ATOMIC_ACQUIRE,
                                         __HIP_MEMORY_SCOPE_AGENT) < need)
                    __builtin_amdgcn_s_sleep(2);
            }
        }
        __syncthreads();

        if constexpr (XRING) {
            // prime the async double-buffer with this chunk's first tile
            // (issued only AFTER the acquire, so data is guaranteed published)
            const int t0 = chunk * CT;
            const f16* g = xin + ((size_t)(t0 & (RING_W - 1)) * BATCH + growA) * HH + hi * 8;
            unsigned l = lds_off_of(sX + ((size_t)(t0 & 1) * BTILE + mrowA) * HH + hi * 8);
            async_ld_b128(l, g);
            async_ld_b128(l + 32, g + 16);
        }

        for (int tt = 0; tt < CT; ++tt) {
            const int t = chunk * CT + tt;

            v16h a[KC];
            if constexpr (XRING) {
                // issue async load of next timestep's tile into the other slot,
                // then wait for this timestep's tile (oldest 2 async ops)
                if (tt + 1 < CT) {
                    const int tn = t + 1;
                    const f16* g = xin + ((size_t)(tn & (RING_W - 1)) * BATCH + growA) * HH + hi * 8;
                    unsigned l = lds_off_of(sX + ((size_t)(tn & 1) * BTILE + mrowA) * HH + hi * 8);
                    async_ld_b128(l, g);
                    async_ld_b128(l + 32, g + 16);
                    WAIT_ASYNC(2);
                } else {
                    WAIT_ASYNC(0);
                }
                const f16* xr = sX + ((size_t)(t & 1) * BTILE + mrowA) * HH;
                V16H tmp;
                tmp.h[0] = *(const v8h*)(xr + hi * 8);
                tmp.h[1] = *(const v8h*)(xr + hi * 8 + 16);
                a[0] = tmp.v;
            } else {
                const f16* xrow = xin + ((size_t)t * BATCH + growA) * IN0;
#pragma unroll
                for (int kc = 0; kc < KXC; ++kc) {
                    V16H tmp;
                    tmp.h[0] = *(const v8h*)(xrow + kc * 32 + hi * 8);
                    tmp.h[1] = *(const v8h*)(xrow + kc * 32 + hi * 8 + 16);
                    a[kc] = tmp.v;
                }
                if (t + 1 < T_STEPS)    // WGP-scope prefetch of next row
                    __builtin_prefetch(xin + ((size_t)(t + 1) * BATCH + growA) * IN0, 0, 3);
            }
            {   // h recurrence part from LDS
                const f16* hrow = sH + mrowA * HH;
                V16H tmp;
                tmp.h[0] = *(const v8h*)(hrow + hi * 8);
                tmp.h[1] = *(const v8h*)(hrow + hi * 8 + 16);
                a[KXC] = tmp.v;
            }
            __syncthreads();   // everyone consumed sH before it is overwritten

            // ---- gates = A x B + bias : 8 N-tiles, KC chained WMMAs each ----
            v8f acc[8];
#pragma unroll
            for (int n = 0; n < 8; ++n) {
                v8f c;
#pragma unroll
                for (int r = 0; r < 8; ++r) c[r] = biasn[n];
#pragma unroll
                for (int kc = 0; kc < KC; ++kc)
                    c = __builtin_amdgcn_wmma_f32_16x16x32_f16(
                            false, a[kc], false, breg[kc][n],
                            (short)0, c, false, false);
                acc[n] = c;
            }

            // ---- LSTM cell update, wave-local (PyTorch i,f,g,o order) ----
            f16 hval[2][8];
#pragma unroll
            for (int j = 0; j < 2; ++j) {
#pragma unroll
                for (int r = 0; r < 8; ++r) {
                    float ig = fast_sig(acc[0 + j][r]);
                    float fg = fast_sig(acc[2 + j][r]);
                    float gg = TANHF  (acc[4 + j][r]);
                    float og = fast_sig(acc[6 + j][r]);
                    float c  = fg * cst[j][r] + ig * gg;
                    cst[j][r]  = c;
                    hval[j][r] = (f16)(og * TANHF(c));
                }
            }

            // ---- scatter h(t) to LDS in row-major (BTILE x HH) ----
#pragma unroll
            for (int j = 0; j < 2; ++j)
#pragma unroll
                for (int r = 0; r < 8; ++r)
                    sH[(rowbaseD + r) * HH + j * 16 + l16] = hval[j][r];
            __syncthreads();

            // ---- cooperative copy h tile LDS -> ring slot (L2-resident) ----
            {
                f16* dst = ring_out +
                    ((size_t)(t & (RING_W - 1)) * BATCH + slice * BTILE) * HH;
#pragma unroll
                for (int q = 0; q < 2; ++q) {
                    int cidx = tid + q * 256;               // 512 x 16B chunks
                    *(v8h*)(dst + cidx * 8) = *(const v8h*)(sH + cidx * 8);
                }
            }
            if (hT_out && t == T_STEPS - 1) {
                f16* dst = hT_out + (size_t)slice * BTILE * HH;
#pragma unroll
                for (int q = 0; q < 2; ++q) {
                    int cidx = tid + q * 256;
                    *(v8h*)(dst + cidx * 8) = *(const v8h*)(sH + cidx * 8);
                }
            }
        }

        // ---- release: publish chunk, free upstream ring space ----
        __threadfence();
        __syncthreads();
        if (tid == 0) {
            __hip_atomic_store(&wprog[myW], chunk + 1, __ATOMIC_RELEASE,
                               __HIP_MEMORY_SCOPE_AGENT);
            if (layer > 0)
                __hip_atomic_store(&rprog[upW], chunk + 1, __ATOMIC_RELEASE,
                                   __HIP_MEMORY_SCOPE_AGENT);
        }
    }
}

// ---------------------------------------------------------------------------
__global__ __launch_bounds__(256, 1)
void lstm_pipeline_kernel(
    const f16* __restrict__ x16, f16* __restrict__ rings, f16* __restrict__ hT,
    int* wprog, int* rprog,
    const float* __restrict__ Wih0, const float* __restrict__ Whh0,
    const float* __restrict__ bih0, const float* __restrict__ bhh0,
    const float* __restrict__ Wih,  const float* __restrict__ Whh,
    const float* __restrict__ bih,  const float* __restrict__ bhh)
{
    __shared__ f16   sW[96 * G4];          // up to (64+32) x 128 f16 weights
    __shared__ float sBias[G4];
    __shared__ f16   sH[BTILE * HH];       // h(t) tile
    __shared__ f16   sX[2 * BTILE * HH];   // async x double-buffer (ring layers)

    const int layer = blockIdx.y;
    const int slice = blockIdx.x;
    f16* ring_out = rings + (size_t)layer * RING_W * BATCH * HH;
    f16* hTo = (layer == NL - 1) ? hT : nullptr;

    if (layer == 0) {
        lstm_layer_pipeline<2, false>(x16, ring_out, hTo, 0, slice, wprog, rprog,
                                      Wih0, Whh0, bih0, bhh0, sW, sBias, sH, sX);
    } else {
        const f16* xin = rings + (size_t)(layer - 1) * RING_W * BATCH * HH;
        lstm_layer_pipeline<1, true>(xin, ring_out, hTo, layer, slice, wprog, rprog,
                                     Wih + (size_t)(layer - 1) * G4 * HH,
                                     Whh + (size_t)(layer - 1) * G4 * HH,
                                     bih + (size_t)(layer - 1) * G4,
                                     bhh + (size_t)(layer - 1) * G4,
                                     sW, sBias, sH, sX);
    }
}

// x: (B, IN0, T) f32  ->  x16: (T, B, IN0) f16
__global__ void convert_x_kernel(const float* __restrict__ x, f16* __restrict__ x16)
{
    size_t i = (size_t)blockIdx.x * blockDim.x + threadIdx.x;
    if (i >= (size_t)T_STEPS * BATCH * IN0) return;
    int    c = (int)(i % IN0);
    size_t r = i / IN0;
    int    b = (int)(r % BATCH);
    int    t = (int)(r / BATCH);
    x16[i] = (f16)x[((size_t)b * IN0 + c) * T_STEPS + t];
}

__global__ void init_flags_kernel(int* wprog, int* rprog)
{
    int i = threadIdx.x;
    if (i < NL * NSLICE) {
        wprog[i] = 0;
        // last layer has no consumer: never apply back-pressure to it
        rprog[i] = (i >= (NL - 1) * NSLICE) ? 0x3FFFFFFF : 0;
    }
}

// out[b][n] = hT[b,:] . Wfc[n,:] + bfc[n]
__global__ void fc_kernel(const f16* __restrict__ hT, const float* __restrict__ Wfc,
                          const float* __restrict__ bfc, float* __restrict__ out)
{
    int b = blockIdx.x * blockDim.x + threadIdx.x;
    if (b >= BATCH) return;
#pragma unroll
    for (int n = 0; n < 4; ++n) {
        float acc = bfc[n];
#pragma unroll
        for (int k = 0; k < HH; ++k)
            acc += (float)hT[b * HH + k] * Wfc[n * HH + k];
        out[b * 4 + n] = acc;
    }
}

// ---------------------------------------------------------------------------
extern "C" void kernel_launch(void* const* d_in, const int* in_sizes, int n_in,
                              void* d_out, int out_size, void* d_ws, size_t ws_size,
                              hipStream_t stream)
{
    const float* x    = (const float*)d_in[0];
    const float* Wih0 = (const float*)d_in[1];
    const float* Whh0 = (const float*)d_in[2];
    const float* bih0 = (const float*)d_in[3];
    const float* bhh0 = (const float*)d_in[4];
    const float* Wih  = (const float*)d_in[5];
    const float* Whh  = (const float*)d_in[6];
    const float* bih  = (const float*)d_in[7];
    const float* bhh  = (const float*)d_in[8];
    const float* Wfc  = (const float*)d_in[9];
    const float* bfc  = (const float*)d_in[10];

    char*  ws  = (char*)d_ws;
    size_t off = 0;
    f16* x16   = (f16*)(ws + off); off += (size_t)T_STEPS * BATCH * IN0 * sizeof(f16);
    f16* rings = (f16*)(ws + off); off += (size_t)NL * RING_W * BATCH * HH * sizeof(f16);
    f16* hT    = (f16*)(ws + off); off += (size_t)BATCH * HH * sizeof(f16);
    off = (off + 255) & ~(size_t)255;
    int* wprog = (int*)(ws + off); off += NL * NSLICE * sizeof(int);
    int* rprog = (int*)(ws + off); off += NL * NSLICE * sizeof(int);

    init_flags_kernel<<<1, 128, 0, stream>>>(wprog, rprog);

    size_t nconv = (size_t)T_STEPS * BATCH * IN0;
    convert_x_kernel<<<(unsigned)((nconv + 255) / 256), 256, 0, stream>>>(x, x16);

    lstm_pipeline_kernel<<<dim3(NSLICE, NL), 256, 0, stream>>>(
        x16, rings, hT, wprog, rprog,
        Wih0, Whh0, bih0, bhh0, Wih, Whh, bih, bhh);

    fc_kernel<<<2, 256, 0, stream>>>(hT, Wfc, bfc, (float*)d_out);
}
